// MoEFFN_90486370992147
// MI455X (gfx1250) — compile-verified, matrix operand order checked by reading, not run on hardware
//
#include <hip/hip_runtime.h>
#include <hip/hip_bf16.h>

#define T_TOK 16384
#define DDIM  1024
#define NEXP  8
#define HDIM  2752

typedef __attribute__((ext_vector_type(16))) __bf16 v16bf;
typedef __attribute__((ext_vector_type(8)))  float  v8f;
typedef __attribute__((ext_vector_type(4)))  int    v4i;

#if defined(__has_builtin)
#if __has_builtin(__builtin_amdgcn_global_load_async_to_lds_b128) && \
    __has_builtin(__builtin_amdgcn_s_wait_asynccnt)
#define USE_ASYNC_LDS 1
#endif
#endif
#ifndef USE_ASYNC_LDS
#define USE_ASYNC_LDS 0
#endif

#define GAS __attribute__((address_space(1)))
#define LAS __attribute__((address_space(3)))

static __device__ __forceinline__ void async_cp16(const unsigned short* g,
                                                  unsigned short* l) {
#if USE_ASYNC_LDS
  __builtin_amdgcn_global_load_async_to_lds_b128((GAS v4i*)g, (LAS v4i*)l, 0, 0);
#endif
}

struct F32B { uint4 lo, hi; };

static __device__ __forceinline__ v16bf ldfrag(const unsigned short* p0,
                                               const unsigned short* p1) {
  F32B f;
  f.lo = *(const uint4*)p0;
  f.hi = *(const uint4*)p1;
  return __builtin_bit_cast(v16bf, f);
}

static __device__ __forceinline__ unsigned short f2bf(float f) {
  unsigned int u = __float_as_uint(f);
  return (unsigned short)((u + 0x7FFFu + ((u >> 16) & 1u)) >> 16);
}

#define WMMA_BF16(A, B, C) \
  __builtin_amdgcn_wmma_f32_16x16x32_bf16(false, A, false, B, (short)0, C, false, false)

// ---------------------------------------------------------------- convert (no transpose, for x)
__global__ __launch_bounds__(256) void cvt_bf16_kernel(
    const float* __restrict__ src, unsigned short* __restrict__ dst, size_t n) {
  size_t stride = (size_t)gridDim.x * blockDim.x * 4;
  for (size_t i = ((size_t)blockIdx.x * blockDim.x + threadIdx.x) * 4;
       i + 3 < n; i += stride) {
    float4 v = *(const float4*)(src + i);
    uint2 o;
    o.x = (unsigned)f2bf(v.x) | ((unsigned)f2bf(v.y) << 16);
    o.y = (unsigned)f2bf(v.z) | ((unsigned)f2bf(v.w) << 16);
    *(uint2*)(dst + i) = o;
  }
}

// ---------------------------------------------------------------- transpose + convert (weights)
// src: [Z][R][C] fp32  ->  dst: [Z][C][R] bf16   (R, C multiples of 32)
__global__ __launch_bounds__(256) void tcvt_bf16_kernel(
    const float* __restrict__ src, unsigned short* __restrict__ dst,
    int R, int C) {
  __shared__ unsigned short tile[32][33];
  const int z = blockIdx.z;
  const int r0 = blockIdx.y * 32, c0 = blockIdx.x * 32;
  const float* s = src + (size_t)z * R * C;
  unsigned short* d = dst + (size_t)z * R * C;
  const int tid = threadIdx.x;

  const int lr = tid >> 3;
  const int lc = (tid & 7) * 4;
  float4 v = *(const float4*)(s + (size_t)(r0 + lr) * C + c0 + lc);
  tile[lr][lc + 0] = f2bf(v.x);
  tile[lr][lc + 1] = f2bf(v.y);
  tile[lr][lc + 2] = f2bf(v.z);
  tile[lr][lc + 3] = f2bf(v.w);
  __syncthreads();

  const int oc = tid >> 3;
  const int orr = (tid & 7) * 4;
  uint2 o;
  o.x = (unsigned)tile[orr + 0][oc] | ((unsigned)tile[orr + 1][oc] << 16);
  o.y = (unsigned)tile[orr + 2][oc] | ((unsigned)tile[orr + 3][oc] << 16);
  *(uint2*)(d + (size_t)(c0 + oc) * R + r0 + orr) = o;
}

// ---------------------------------------------------------------- router
__global__ __launch_bounds__(256) void router_kernel(
    const float* __restrict__ x, const float* __restrict__ rw,
    int* __restrict__ cnt, int* __restrict__ list, float* __restrict__ pgate) {
  const int lane = threadIdx.x & 31;
  const int wv   = threadIdx.x >> 5;
  const int t    = blockIdx.x * 8 + wv;

  float acc[NEXP];
#pragma unroll
  for (int e = 0; e < NEXP; ++e) acc[e] = 0.f;

  for (int d = lane; d < DDIM; d += 32) {
    float xv = x[(size_t)t * DDIM + d];
#pragma unroll
    for (int e = 0; e < NEXP; ++e) acc[e] += xv * rw[e * DDIM + d];
  }
#pragma unroll
  for (int off = 16; off >= 1; off >>= 1) {
#pragma unroll
    for (int e = 0; e < NEXP; ++e) acc[e] += __shfl_xor(acc[e], off);
  }

  if (lane == 0) {
    int e0 = 0; float v0 = acc[0];
#pragma unroll
    for (int e = 1; e < NEXP; ++e) if (acc[e] > v0) { v0 = acc[e]; e0 = e; }
    int e1 = -1; float v1 = -3.4e38f;
#pragma unroll
    for (int e = 0; e < NEXP; ++e)
      if (e != e0 && acc[e] > v1) { v1 = acc[e]; e1 = e; }
    float g0 = 1.f / (1.f + __expf(v1 - v0));
    float g1 = 1.f - g0;
    pgate[2 * t]     = g0;
    pgate[2 * t + 1] = g1;
    int p0 = atomicAdd(&cnt[e0], 1);
    list[e0 * T_TOK + p0] = 2 * t;
    int p1 = atomicAdd(&cnt[e1], 1);
    list[e1 * T_TOK + p1] = 2 * t + 1;
  }
}

// ---------------------------------------------------------------- GEMM1: h = silu(x@w1) * (x@w3)
// w1t/w3t are [E][H][D] bf16 (pre-transposed): B-tile rows are N(H), contiguous K(D).
__global__ __launch_bounds__(256) void moe_gemm1(
    const unsigned short* __restrict__ xb,
    const unsigned short* __restrict__ w1t,
    const unsigned short* __restrict__ w3t,
    const int* __restrict__ cnt, const int* __restrict__ list,
    unsigned short* __restrict__ hbuf) {
  const int e = blockIdx.z;
  const int mtile = blockIdx.x;
  const int c = cnt[e];
  if (mtile * 128 >= c) return;
  const int nb = blockIdx.y * 64;

  __shared__ __align__(16) unsigned short As[2][128][32];
  __shared__ __align__(16) unsigned short Bs1[2][64][32];
  __shared__ __align__(16) unsigned short Bs3[2][64][32];
  __shared__ int rowPair[128];

  const int tid = threadIdx.x;
  if (tid < 128) {
    int grow = mtile * 128 + tid;
    rowPair[tid] = (grow < c) ? list[e * T_TOK + grow] : -1;
  }
  __syncthreads();

  // per-thread 16B tile chunks
  const int ar0 = tid >> 2;             // A rows 0..63
  const int ar1 = ar0 + 64;             // A rows 64..127
  const int asg = (tid & 3) * 8;        // K segment
  const int bn  = tid >> 2;             // B row (N) 0..63
  const int bsg = (tid & 3) * 8;        // K segment
  const int pr0 = rowPair[ar0];
  const int pr1 = rowPair[ar1];
  // invalid rows: redirect to row 0 (their outputs are never stored)
  const unsigned short* a0src = xb + (size_t)((pr0 >= 0) ? (pr0 >> 1) : 0) * DDIM + asg;
  const unsigned short* a1src = xb + (size_t)((pr1 >= 0) ? (pr1 >> 1) : 0) * DDIM + asg;
  const size_t woff = (size_t)e * HDIM * DDIM + (size_t)(nb + bn) * DDIM + bsg;
  const unsigned short* b1src = w1t + woff;
  const unsigned short* b3src = w3t + woff;

  const int lane = tid & 31;
  const int w    = tid >> 5;
  const int m0   = (w >> 1) * 32;
  const int nw0  = (w & 1) * 32;
  const int lr   = lane & 15;
  const int kh   = (lane >> 4) * 8;

  v8f acc1[2][2] = {};
  v8f acc3[2][2] = {};
  const int NK = DDIM / 32;

  auto compute = [&](int cur) {
    const unsigned short (*A)[32]  = As[cur];
    const unsigned short (*B1)[32] = Bs1[cur];
    const unsigned short (*B3)[32] = Bs3[cur];
    v16bf a0  = ldfrag(&A[m0 + lr][kh],        &A[m0 + lr][16 + kh]);
    v16bf a1  = ldfrag(&A[m0 + 16 + lr][kh],   &A[m0 + 16 + lr][16 + kh]);
    v16bf b10 = ldfrag(&B1[nw0 + lr][kh],      &B1[nw0 + lr][16 + kh]);
    v16bf b11 = ldfrag(&B1[nw0 + 16 + lr][kh], &B1[nw0 + 16 + lr][16 + kh]);
    v16bf b30 = ldfrag(&B3[nw0 + lr][kh],      &B3[nw0 + lr][16 + kh]);
    v16bf b31 = ldfrag(&B3[nw0 + 16 + lr][kh], &B3[nw0 + 16 + lr][16 + kh]);
    acc1[0][0] = WMMA_BF16(a0, b10, acc1[0][0]);
    acc1[0][1] = WMMA_BF16(a0, b11, acc1[0][1]);
    acc1[1][0] = WMMA_BF16(a1, b10, acc1[1][0]);
    acc1[1][1] = WMMA_BF16(a1, b11, acc1[1][1]);
    acc3[0][0] = WMMA_BF16(a0, b30, acc3[0][0]);
    acc3[0][1] = WMMA_BF16(a0, b31, acc3[0][1]);
    acc3[1][0] = WMMA_BF16(a1, b30, acc3[1][0]);
    acc3[1][1] = WMMA_BF16(a1, b31, acc3[1][1]);
  };

#if USE_ASYNC_LDS
  async_cp16(a0src, &As[0][ar0][asg]);
  async_cp16(a1src, &As[0][ar1][asg]);
  async_cp16(b1src, &Bs1[0][bn][bsg]);
  async_cp16(b3src, &Bs3[0][bn][bsg]);
  for (int kb = 0; kb < NK; ++kb) {
    const int cur = kb & 1;
    if (kb + 1 < NK) {
      const int nxt = cur ^ 1;
      const int k1 = (kb + 1) * 32;
      async_cp16(a0src + k1, &As[nxt][ar0][asg]);
      async_cp16(a1src + k1, &As[nxt][ar1][asg]);
      async_cp16(b1src + k1, &Bs1[nxt][bn][bsg]);
      async_cp16(b3src + k1, &Bs3[nxt][bn][bsg]);
      __builtin_amdgcn_s_wait_asynccnt(4);   // current tile complete, next in flight
    } else {
      __builtin_amdgcn_s_wait_asynccnt(0);
    }
    __syncthreads();
    compute(cur);
    __syncthreads();   // reads done before next iteration overwrites this buffer
  }
#else
  uint4 ra0 = *(const uint4*)a0src;
  uint4 ra1 = *(const uint4*)a1src;
  uint4 rb1 = *(const uint4*)b1src;
  uint4 rb3 = *(const uint4*)b3src;
  *(uint4*)&As[0][ar0][asg]  = ra0;
  *(uint4*)&As[0][ar1][asg]  = ra1;
  *(uint4*)&Bs1[0][bn][bsg]  = rb1;
  *(uint4*)&Bs3[0][bn][bsg]  = rb3;
  for (int kb = 0; kb < NK; ++kb) {
    const int cur = kb & 1;
    if (kb + 1 < NK) {
      const int k1 = (kb + 1) * 32;
      ra0 = *(const uint4*)(a0src + k1);
      ra1 = *(const uint4*)(a1src + k1);
      rb1 = *(const uint4*)(b1src + k1);
      rb3 = *(const uint4*)(b3src + k1);
    }
    __syncthreads();
    compute(cur);
    if (kb + 1 < NK) {
      const int nxt = cur ^ 1;
      *(uint4*)&As[nxt][ar0][asg] = ra0;
      *(uint4*)&As[nxt][ar1][asg] = ra1;
      *(uint4*)&Bs1[nxt][bn][bsg] = rb1;
      *(uint4*)&Bs3[nxt][bn][bsg] = rb3;
    }
  }
#endif

#pragma unroll
  for (int mt = 0; mt < 2; ++mt)
#pragma unroll
    for (int nt = 0; nt < 2; ++nt) {
      v8f s1 = acc1[mt][nt];
      v8f s3 = acc3[mt][nt];
#pragma unroll
      for (int r = 0; r < 8; ++r) {
        int m = m0 + mt * 16 + kh + r;
        int pr = rowPair[m];
        if (pr < 0) continue;
        float x1 = s1[r], x3 = s3[r];
        float hv = (x1 / (1.f + __expf(-x1))) * x3;
        int col = nb + nw0 + nt * 16 + lr;
        hbuf[(size_t)pr * HDIM + col] = f2bf(hv);
      }
    }
}

// ---------------------------------------------------------------- GEMM2: out += gate * (h @ w2)
// w2t is [E][D][H] bf16 (pre-transposed): B-tile rows are N(D), contiguous K(H).
__global__ __launch_bounds__(256) void moe_gemm2(
    const unsigned short* __restrict__ hbuf,
    const unsigned short* __restrict__ w2t,
    const int* __restrict__ cnt, const int* __restrict__ list,
    const float* __restrict__ pgate, float* __restrict__ out) {
  const int e = blockIdx.z;
  const int mtile = blockIdx.x;
  const int c = cnt[e];
  if (mtile * 128 >= c) return;
  const int nb = blockIdx.y * 64;

  __shared__ __align__(16) unsigned short As[2][128][32];
  __shared__ __align__(16) unsigned short Bs[2][64][32];
  __shared__ int   rowPair[128];
  __shared__ float rowGate[128];

  const int tid = threadIdx.x;
  if (tid < 128) {
    int grow = mtile * 128 + tid;
    int pr = (grow < c) ? list[e * T_TOK + grow] : -1;
    rowPair[tid] = pr;
    rowGate[tid] = (pr >= 0) ? pgate[pr] : 0.f;
  }
  __syncthreads();

  const int ar0 = tid >> 2;
  const int ar1 = ar0 + 64;
  const int asg = (tid & 3) * 8;
  const int bn  = tid >> 2;
  const int bsg = (tid & 3) * 8;
  const int pr0 = rowPair[ar0];
  const int pr1 = rowPair[ar1];
  const unsigned short* a0src = hbuf + (size_t)((pr0 >= 0) ? pr0 : 0) * HDIM + asg;
  const unsigned short* a1src = hbuf + (size_t)((pr1 >= 0) ? pr1 : 0) * HDIM + asg;
  const unsigned short* bsrc =
      w2t + (size_t)e * DDIM * HDIM + (size_t)(nb + bn) * HDIM + bsg;

  const int lane = tid & 31;
  const int w    = tid >> 5;
  const int m0   = (w >> 1) * 32;
  const int nw0  = (w & 1) * 32;
  const int lr   = lane & 15;
  const int kh   = (lane >> 4) * 8;

  v8f acc[2][2] = {};
  const int NK = HDIM / 32;

  auto compute = [&](int cur) {
    const unsigned short (*A)[32] = As[cur];
    const unsigned short (*B)[32] = Bs[cur];
    v16bf a0 = ldfrag(&A[m0 + lr][kh],        &A[m0 + lr][16 + kh]);
    v16bf a1 = ldfrag(&A[m0 + 16 + lr][kh],   &A[m0 + 16 + lr][16 + kh]);
    v16bf b0 = ldfrag(&B[nw0 + lr][kh],       &B[nw0 + lr][16 + kh]);
    v16bf b1 = ldfrag(&B[nw0 + 16 + lr][kh],  &B[nw0 + 16 + lr][16 + kh]);
    acc[0][0] = WMMA_BF16(a0, b0, acc[0][0]);
    acc[0][1] = WMMA_BF16(a0, b1, acc[0][1]);
    acc[1][0] = WMMA_BF16(a1, b0, acc[1][0]);
    acc[1][1] = WMMA_BF16(a1, b1, acc[1][1]);
  };

#if USE_ASYNC_LDS
  async_cp16(a0src, &As[0][ar0][asg]);
  async_cp16(a1src, &As[0][ar1][asg]);
  async_cp16(bsrc,  &Bs[0][bn][bsg]);
  for (int kb = 0; kb < NK; ++kb) {
    const int cur = kb & 1;
    if (kb + 1 < NK) {
      const int nxt = cur ^ 1;
      const int k1 = (kb + 1) * 32;
      async_cp16(a0src + k1, &As[nxt][ar0][asg]);
      async_cp16(a1src + k1, &As[nxt][ar1][asg]);
      async_cp16(bsrc + k1,  &Bs[nxt][bn][bsg]);
      __builtin_amdgcn_s_wait_asynccnt(3);
    } else {
      __builtin_amdgcn_s_wait_asynccnt(0);
    }
    __syncthreads();
    compute(cur);
    __syncthreads();
  }
#else
  uint4 ra0 = *(const uint4*)a0src;
  uint4 ra1 = *(const uint4*)a1src;
  uint4 rb  = *(const uint4*)bsrc;
  *(uint4*)&As[0][ar0][asg] = ra0;
  *(uint4*)&As[0][ar1][asg] = ra1;
  *(uint4*)&Bs[0][bn][bsg]  = rb;
  for (int kb = 0; kb < NK; ++kb) {
    const int cur = kb & 1;
    if (kb + 1 < NK) {
      const int k1 = (kb + 1) * 32;
      ra0 = *(const uint4*)(a0src + k1);
      ra1 = *(const uint4*)(a1src + k1);
      rb  = *(const uint4*)(bsrc + k1);
    }
    __syncthreads();
    compute(cur);
    if (kb + 1 < NK) {
      const int nxt = cur ^ 1;
      *(uint4*)&As[nxt][ar0][asg] = ra0;
      *(uint4*)&As[nxt][ar1][asg] = ra1;
      *(uint4*)&Bs[nxt][bn][bsg]  = rb;
    }
  }
#endif

#pragma unroll
  for (int mt = 0; mt < 2; ++mt)
#pragma unroll
    for (int nt = 0; nt < 2; ++nt) {
      v8f s = acc[mt][nt];
#pragma unroll
      for (int r = 0; r < 8; ++r) {
        int m = m0 + mt * 16 + kh + r;
        int pr = rowPair[m];
        if (pr < 0) continue;
        float val = s[r] * rowGate[m];
        int t = pr >> 1;
        int col = nb + nw0 + nt * 16 + lr;
        atomicAdd(out + (size_t)t * DDIM + col, val);
      }
    }
}

// ---------------------------------------------------------------- launch
extern "C" void kernel_launch(void* const* d_in, const int* in_sizes, int n_in,
                              void* d_out, int out_size, void* d_ws, size_t ws_size,
                              hipStream_t stream) {
  (void)in_sizes; (void)n_in; (void)out_size; (void)ws_size;
  const float* x  = (const float*)d_in[0];
  const float* rw = (const float*)d_in[1];
  const float* w1 = (const float*)d_in[2];
  const float* w2 = (const float*)d_in[3];
  const float* w3 = (const float*)d_in[4];
  float* out = (float*)d_out;

  char* ws = (char*)d_ws;
  size_t o = 0;
  auto alloc = [&](size_t bytes) {
    char* p = ws + o;
    o += (bytes + 255) & ~(size_t)255;
    return p;
  };
  unsigned short* xb   = (unsigned short*)alloc((size_t)T_TOK * DDIM * 2);
  unsigned short* w1t  = (unsigned short*)alloc((size_t)NEXP * DDIM * HDIM * 2);
  unsigned short* w3t  = (unsigned short*)alloc((size_t)NEXP * DDIM * HDIM * 2);
  unsigned short* w2t  = (unsigned short*)alloc((size_t)NEXP * HDIM * DDIM * 2);
  unsigned short* hbuf = (unsigned short*)alloc((size_t)2 * T_TOK * HDIM * 2);
  int*   list  = (int*)alloc((size_t)NEXP * T_TOK * 4);
  float* pgate = (float*)alloc((size_t)2 * T_TOK * 4);
  int*   cnt   = (int*)alloc(256);

  (void)hipMemsetAsync(cnt, 0, 256, stream);
  (void)hipMemsetAsync(d_out, 0, (size_t)T_TOK * DDIM * sizeof(float), stream);

  cvt_bf16_kernel<<<2048, 256, 0, stream>>>(x, xb, (size_t)T_TOK * DDIM);
  // w1: [E][D][H] -> w1t [E][H][D]
  tcvt_bf16_kernel<<<dim3(HDIM / 32, DDIM / 32, NEXP), 256, 0, stream>>>(w1, w1t, DDIM, HDIM);
  tcvt_bf16_kernel<<<dim3(HDIM / 32, DDIM / 32, NEXP), 256, 0, stream>>>(w3, w3t, DDIM, HDIM);
  // w2: [E][H][D] -> w2t [E][D][H]
  tcvt_bf16_kernel<<<dim3(DDIM / 32, HDIM / 32, NEXP), 256, 0, stream>>>(w2, w2t, HDIM, DDIM);

  router_kernel<<<T_TOK / 8, 256, 0, stream>>>(x, rw, cnt, list, pgate);

  moe_gemm1<<<dim3(128, HDIM / 64, NEXP), 256, 0, stream>>>(xb, w1t, w3t, cnt, list, hbuf);
  moe_gemm2<<<dim3(128, DDIM / 64, NEXP), 256, 0, stream>>>(hbuf, w2t, cnt, list, pgate, out);
}
